// MiniMaxDecoderLayer_59803124630221
// MI455X (gfx1250) — compile-verified, matrix operand order checked by reading, not run on hardware
//
#include <hip/hip_runtime.h>
#include <hip/hip_bf16.h>

#define T_TOK 8192
#define HDIM  2048
#define IDIM  1408
#define NEXP  8

typedef __attribute__((ext_vector_type(16))) __bf16 v16bf;
typedef __attribute__((ext_vector_type(8)))  __bf16 v8bf;
typedef __attribute__((ext_vector_type(4)))  __bf16 v4bf;
typedef __attribute__((ext_vector_type(8)))  float  v8f;

union FragU { v16bf v; v8bf h[2]; };

// ---------------- zero output + expert counters ----------------
__global__ void k_zero(float4* __restrict__ out4, int n4, int* __restrict__ cnt) {
  int i = blockIdx.x * blockDim.x + threadIdx.x;
  int stride = gridDim.x * blockDim.x;
  float4 z; z.x = z.y = z.z = z.w = 0.f;
  for (; i < n4; i += stride) out4[i] = z;
  if (blockIdx.x == 0 && threadIdx.x < NEXP) cnt[threadIdx.x] = 0;
}

// ---------------- fp32 -> bf16 elementwise (activations) ----------------
__global__ void k_cvt_bf16(const float4* __restrict__ in, v4bf* __restrict__ outv, int n4) {
  int i = blockIdx.x * blockDim.x + threadIdx.x;
  int stride = gridDim.x * blockDim.x;
  for (; i < n4; i += stride) {
    float4 v = in[i];
    v4bf o = { (__bf16)v.x, (__bf16)v.y, (__bf16)v.z, (__bf16)v.w };
    outv[i] = o;
  }
}

// ------- fp32 -> bf16 transpose: src[e][R][C] -> dst[e][C][R] -------
__global__ void k_transpose_bf16(const float* __restrict__ src, __bf16* __restrict__ dst,
                                 int R, int C) {
  __shared__ float tile[32][33];
  int e = blockIdx.z;
  size_t base = (size_t)e * R * C;
  int c0 = blockIdx.x * 32, r0 = blockIdx.y * 32;
  int tx = threadIdx.x & 31, ty = threadIdx.x >> 5;
  #pragma unroll
  for (int i = 0; i < 32; i += 8)
    tile[ty + i][tx] = src[base + (size_t)(r0 + ty + i) * C + (c0 + tx)];
  __syncthreads();
  #pragma unroll
  for (int i = 0; i < 32; i += 8)
    dst[base + (size_t)(c0 + ty + i) * R + (r0 + tx)] = (__bf16)tile[tx][ty + i];
}

// ---------------- router: softmax + top-2 + compaction ----------------
__global__ void k_router(const float* __restrict__ x, const float* __restrict__ gw,
                         int* __restrict__ cnt, int* __restrict__ tokl,
                         float* __restrict__ wtl) {
  int lane = threadIdx.x & 31;
  int wv   = threadIdx.x >> 5;
  int t = blockIdx.x * 8 + wv;
  const float* xr = x + (size_t)t * HDIM;
  float acc[NEXP];
  #pragma unroll
  for (int e = 0; e < NEXP; e++) acc[e] = 0.f;
  for (int c = lane; c < HDIM; c += 32) {
    float xv = xr[c];
    #pragma unroll
    for (int e = 0; e < NEXP; e++) acc[e] = fmaf(xv, gw[e * HDIM + c], acc[e]);
  }
  #pragma unroll
  for (int e = 0; e < NEXP; e++) {
    #pragma unroll
    for (int s = 16; s > 0; s >>= 1) acc[e] += __shfl_xor(acc[e], s, 32);
  }
  if (lane == 0) {
    float mx = acc[0];
    #pragma unroll
    for (int e = 1; e < NEXP; e++) mx = fmaxf(mx, acc[e]);
    float p[NEXP];
    #pragma unroll
    for (int e = 0; e < NEXP; e++) p[e] = __expf(acc[e] - mx);
    int i0 = 0;
    #pragma unroll
    for (int e = 1; e < NEXP; e++) if (p[e] > p[i0]) i0 = e;
    int i1 = (i0 == 0) ? 1 : 0;
    #pragma unroll
    for (int e = 0; e < NEXP; e++) if (e != i0 && p[e] > p[i1]) i1 = e;
    float s2 = p[i0] + p[i1];
    float w0 = p[i0] / s2, w1 = p[i1] / s2;
    int q0 = atomicAdd(&cnt[i0], 1);
    tokl[i0 * T_TOK + q0] = t; wtl[i0 * T_TOK + q0] = w0;
    int q1 = atomicAdd(&cnt[i1], 1);
    tokl[i1 * T_TOK + q1] = t; wtl[i1 * T_TOK + q1] = w1;
  }
}

// ---------------- exclusive scan over 8 expert counts ----------------
__global__ void k_scan(const int* __restrict__ cnt, int* __restrict__ offs) {
  int s = 0;
  for (int e = 0; e < NEXP; e++) { offs[e] = s; s += cnt[e]; }
}

// ------- gate/up GEMMs + SiLU: BM=64 (4 sub-tiles/wave), barrier-free K loop -------
__global__ void __launch_bounds__(256)
k_ffn_up(const __bf16* __restrict__ xb, const __bf16* __restrict__ wgt,
         const __bf16* __restrict__ wut, const int* __restrict__ cnt,
         const int* __restrict__ offs, const int* __restrict__ tokl,
         __bf16* __restrict__ act) {
  int e = blockIdx.z;
  int count = cnt[e];
  int m0 = blockIdx.y * 64;
  if (m0 >= count) return;                    // uniform per block
  int off_e = offs[e];

  __shared__ int stok[64];
  __shared__ __bf16 Cs[8][16][24];            // per-wave epilogue tile (48B rows, 16B aligned)

  int tid = threadIdx.x, lane = tid & 31, wv = tid >> 5;
  if (tid < 64) {
    int p = m0 + tid;
    stok[tid] = tokl[e * T_TOK + ((p < count) ? p : 0)];
  }
  __syncthreads();

  int nn = lane & 15, kh = lane >> 4;
  int n0 = blockIdx.x * 128 + wv * 16;
  const __bf16* bgp = wgt + (size_t)e * IDIM * HDIM + (size_t)(n0 + nn) * HDIM + kh * 8;
  const __bf16* bup = wut + (size_t)e * IDIM * HDIM + (size_t)(n0 + nn) * HDIM + kh * 8;

  const __bf16* arow[4];
  #pragma unroll
  for (int s = 0; s < 4; s++)
    arow[s] = xb + (size_t)stok[s * 16 + nn] * HDIM + kh * 8;   // gathered rows, direct global

  v8f cg[4], cu[4];
  #pragma unroll
  for (int s = 0; s < 4; s++) {
    cg[s] = (v8f){0.f,0.f,0.f,0.f,0.f,0.f,0.f,0.f};
    cu[s] = (v8f){0.f,0.f,0.f,0.f,0.f,0.f,0.f,0.f};
  }

  for (int k0 = 0; k0 < HDIM; k0 += 32) {
    FragU bg, bu;
    bg.h[0] = *(const v8bf*)(bgp + k0);
    bg.h[1] = *(const v8bf*)(bgp + k0 + 16);
    bu.h[0] = *(const v8bf*)(bup + k0);
    bu.h[1] = *(const v8bf*)(bup + k0 + 16);
    #pragma unroll
    for (int s = 0; s < 4; s++) {
      FragU a;
      a.h[0] = *(const v8bf*)(arow[s] + k0);
      a.h[1] = *(const v8bf*)(arow[s] + k0 + 16);
      cg[s] = __builtin_amdgcn_wmma_f32_16x16x32_bf16(false, a.v, false, bg.v, (short)0, cg[s], false, false);
      cu[s] = __builtin_amdgcn_wmma_f32_16x16x32_bf16(false, a.v, false, bu.v, (short)0, cu[s], false, false);
    }
  }

  // SiLU(g)*u epilogue per sub-tile, staged via LDS for coalesced bf16 stores
  int row = lane >> 1, colb = (lane & 1) * 8;
  #pragma unroll
  for (int s = 0; s < 4; s++) {
    #pragma unroll
    for (int r = 0; r < 8; r++) {
      float g = cg[s][r], u = cu[s][r];
      Cs[wv][kh * 8 + r][nn] = (__bf16)((g / (1.f + __expf(-g))) * u);
    }
    __syncthreads();
    int p = m0 + s * 16 + row;
    if (p < count)
      *(v8bf*)(act + (size_t)(off_e + p) * IDIM + n0 + colb) = *(const v8bf*)&Cs[wv][row][colb];
    __syncthreads();
  }
}

// ------- down GEMM + weighted scatter-add: BM=64, barrier-free K loop -------
__global__ void __launch_bounds__(256)
k_ffn_down(const __bf16* __restrict__ act, const __bf16* __restrict__ wdt,
           const int* __restrict__ cnt, const int* __restrict__ offs,
           const int* __restrict__ tokl, const float* __restrict__ wtl,
           float* __restrict__ out) {
  int e = blockIdx.z;
  int count = cnt[e];
  int m0 = blockIdx.y * 64;
  if (m0 >= count) return;
  int off_e = offs[e];

  __shared__ float Cs[8][16][17];
  __shared__ int   stok[64];
  __shared__ float swt[64];

  int tid = threadIdx.x, lane = tid & 31, wv = tid >> 5;
  if (tid < 64) {
    int p = m0 + tid;
    bool v = p < count;
    stok[tid] = v ? tokl[e * T_TOK + p] : 0;
    swt[tid]  = v ? wtl[e * T_TOK + p] : 0.f;
  }
  __syncthreads();

  int nn = lane & 15, kh = lane >> 4;
  int n0 = blockIdx.x * 128 + wv * 16;
  const __bf16* bp = wdt + (size_t)e * HDIM * IDIM + (size_t)(n0 + nn) * IDIM + kh * 8;
  const __bf16* ap[4];
  #pragma unroll
  for (int s = 0; s < 4; s++)
    ap[s] = act + (size_t)(off_e + m0 + s * 16 + nn) * IDIM + kh * 8;   // contiguous rows

  v8f c[4];
  #pragma unroll
  for (int s = 0; s < 4; s++) c[s] = (v8f){0.f,0.f,0.f,0.f,0.f,0.f,0.f,0.f};

  for (int k0 = 0; k0 < IDIM; k0 += 32) {
    FragU b;
    b.h[0] = *(const v8bf*)(bp + k0);
    b.h[1] = *(const v8bf*)(bp + k0 + 16);
    #pragma unroll
    for (int s = 0; s < 4; s++) {
      FragU a;
      a.h[0] = *(const v8bf*)(ap[s] + k0);
      a.h[1] = *(const v8bf*)(ap[s] + k0 + 16);
      c[s] = __builtin_amdgcn_wmma_f32_16x16x32_bf16(false, a.v, false, b.v, (short)0, c[s], false, false);
    }
  }

  int row = lane >> 1, colb = (lane & 1) * 8;
  #pragma unroll
  for (int s = 0; s < 4; s++) {
    #pragma unroll
    for (int r = 0; r < 8; r++) Cs[wv][kh * 8 + r][nn] = c[s][r];
    __syncthreads();
    int p = m0 + s * 16 + row;
    if (p < count) {                     // padded rows never touch out (blocks NaN garbage)
      float w   = swt[s * 16 + row];
      float* op = out + (size_t)stok[s * 16 + row] * HDIM + n0 + colb;
      #pragma unroll
      for (int j = 0; j < 8; j++) atomicAdd(op + j, Cs[wv][row][colb + j] * w);
    }
    __syncthreads();
  }
}

// ---------------------------- host launcher ----------------------------
extern "C" void kernel_launch(void* const* d_in, const int* in_sizes, int n_in,
                              void* d_out, int out_size, void* d_ws, size_t ws_size,
                              hipStream_t stream) {
  (void)in_sizes; (void)n_in; (void)out_size; (void)ws_size;
  const float* x  = (const float*)d_in[0];   // [T, H]
  const float* gw = (const float*)d_in[1];   // [E, H]
  const float* wg = (const float*)d_in[2];   // [E, H, I]
  const float* wu = (const float*)d_in[3];   // [E, H, I]
  const float* wd = (const float*)d_in[4];   // [E, I, H]
  float* out = (float*)d_out;                // [T, H]

  char* ws = (char*)d_ws;
  size_t o = 0;
  auto carve = [&](size_t bytes) { char* p = ws + o; o = (o + bytes + 255) & ~(size_t)255; return p; };
  const size_t WBYTES = (size_t)NEXP * IDIM * HDIM * sizeof(__bf16);
  __bf16* wgt  = (__bf16*)carve(WBYTES);                                     // [E][I][H]
  __bf16* wut  = (__bf16*)carve(WBYTES);                                     // [E][I][H]
  __bf16* wdt  = (__bf16*)carve(WBYTES);                                     // [E][H][I]
  __bf16* xb   = (__bf16*)carve((size_t)T_TOK * HDIM * sizeof(__bf16));      // [T][H]
  __bf16* actb = (__bf16*)carve((size_t)(2 * T_TOK + 64) * IDIM * sizeof(__bf16));
  int*    cnt  = (int*)carve(NEXP * sizeof(int));
  int*    offs = (int*)carve(NEXP * sizeof(int));
  int*    tokl = (int*)carve((size_t)NEXP * T_TOK * sizeof(int));
  float*  wtl  = (float*)carve((size_t)NEXP * T_TOK * sizeof(float));

  k_zero<<<4096, 256, 0, stream>>>((float4*)out, (T_TOK * HDIM) / 4, cnt);
  k_cvt_bf16<<<4096, 256, 0, stream>>>((const float4*)x, (v4bf*)xb, (T_TOK * HDIM) / 4);
  k_transpose_bf16<<<dim3(IDIM / 32, HDIM / 32, NEXP), 256, 0, stream>>>(wg, wgt, HDIM, IDIM);
  k_transpose_bf16<<<dim3(IDIM / 32, HDIM / 32, NEXP), 256, 0, stream>>>(wu, wut, HDIM, IDIM);
  k_transpose_bf16<<<dim3(HDIM / 32, IDIM / 32, NEXP), 256, 0, stream>>>(wd, wdt, IDIM, HDIM);
  k_router<<<T_TOK / 8, 256, 0, stream>>>(x, gw, cnt, tokl, wtl);
  k_scan<<<1, 1, 0, stream>>>(cnt, offs);
  k_ffn_up<<<dim3(IDIM / 128, T_TOK / 64, NEXP), 256, 0, stream>>>(xb, wgt, wut, cnt, offs, tokl, actb);
  k_ffn_down<<<dim3(HDIM / 128, T_TOK / 64, NEXP), 256, 0, stream>>>(actb, wdt, cnt, offs, tokl, wtl, out);
}
